// HybridCnnLnn_6493990551899
// MI455X (gfx1250) — matrix operand order repaired and checked by reading, not compiled, not run on hardware
//
#include <hip/hip_runtime.h>
#include <hip/hip_bf16.h>

// ---------- WMMA vector types (per CDNA5_HIP.md, codegen-verified) ----------
typedef __attribute__((ext_vector_type(16))) _Float16 v16h;
typedef __attribute__((ext_vector_type(8)))  _Float16 v8h;
typedef __attribute__((ext_vector_type(8)))  float    v8f;

// A-fragment (16x32 f16, MxK): lane m = l&15, half = l>>4.
// elements e=0..7 -> K = 8*half + e (contiguous), e=8..15 -> K = 8*half + 16 + (e-8).
__device__ __forceinline__ v16h load_a_frag(const _Float16* rowp /* &As[m][kb] */, int lane) {
    const int h = (lane >> 4) & 1;
    v8h lo = *(const v8h*)(rowp + h * 8);
    v8h hi = *(const v8h*)(rowp + h * 8 + 16);
    v16h a;
#pragma unroll
    for (int e = 0; e < 8; ++e) { a[e] = lo[e]; a[e + 8] = hi[e]; }
    return a;
}

// B-fragment (32x16 f16, KxN): lane n = l&15, half = l>>4; same K order as A-fragment.
__device__ __forceinline__ v16h load_b_frag(const _Float16* colp /* &Bs[kb][nb + (lane&15)] */, int ld, int lane) {
    const int h = (lane >> 4) & 1;
    v16h b;
#pragma unroll
    for (int e = 0; e < 8; ++e) { b[e] = colp[(h * 8 + e) * ld]; b[e + 8] = colp[(h * 8 + 16 + e) * ld]; }
    return b;
}

// ======================= Kernel 1: conv1 + relu + pool ======================
// x:[512][256][24] f32, w1:[64][24][5], b1:[64]  ->  h1:[512][128][64] f16
// GEMM view: M=B*T (tile 128 rows), K=120 (pad 128), N=64.
__global__ __launch_bounds__(256) void k_conv1(const float* __restrict__ x,
                                               const float* __restrict__ w1,
                                               const float* __restrict__ b1,
                                               _Float16* __restrict__ h1) {
    __shared__ _Float16 smem[128 * 128 + 128 * 64];  // As + Bs (Cs aliases Bs), 48 KB
    _Float16* As = smem;             // [128][128]
    _Float16* Bs = smem + 128 * 128; // [128][64]
    const int tid = threadIdx.x;
    const int b  = blockIdx.x >> 1;
    const int t0 = (blockIdx.x & 1) * 128;

    // Weights: Bs[k=tap*24+f][c] = w1[c*120 + f*5 + tap]
    for (int idx = tid; idx < 128 * 64; idx += 256) {
        const int k = idx >> 6, c = idx & 63;
        _Float16 v = (_Float16)0.f;
        if (k < 120) { const int tap = k / 24, f = k - tap * 24; v = (_Float16)w1[c * 120 + f * 5 + tap]; }
        Bs[k * 64 + c] = v;
    }
    // im2col: As[row][k] = x[b][t0+row+tap-2][f]
    for (int idx = tid; idx < 128 * 128; idx += 256) {
        const int row = idx >> 7, k = idx & 127;
        _Float16 v = (_Float16)0.f;
        if (k < 120) {
            const int tap = k / 24, f = k - tap * 24;
            const int tt = t0 + row + tap - 2;
            if (tt >= 0 && tt < 256) v = (_Float16)x[(b * 256 + tt) * 24 + f];
        }
        As[row * 128 + k] = v;
    }
    __syncthreads();

    const int lane = tid & 31, wv = tid >> 5;
    const int mrow = wv * 16 + (lane & 15);
    v8f acc[4] = {};
#pragma unroll
    for (int s = 0; s < 4; ++s) {
        const int kb = s * 32;
        v16h a = load_a_frag(As + mrow * 128 + kb, lane);
#pragma unroll
        for (int nt = 0; nt < 4; ++nt) {
            v16h bf = load_b_frag(Bs + kb * 64 + nt * 16 + (lane & 15), 64, lane);
            acc[nt] = __builtin_amdgcn_wmma_f32_16x16x32_f16(false, a, false, bf,
                                                             (short)0, acc[nt], false, false);
        }
    }
    __syncthreads();                 // Bs no longer needed -> reuse as Cs
    _Float16* Cs = Bs;               // [128][64]
    const int colh = lane & 15, rh = lane >> 4;
#pragma unroll
    for (int nt = 0; nt < 4; ++nt) {
        const int c = nt * 16 + colh;
        const float bias = b1[c];
#pragma unroll
        for (int r = 0; r < 8; ++r) {
            const int m = wv * 16 + r + 8 * rh;
            const float val = acc[nt][r] + bias;
            Cs[m * 64 + c] = (_Float16)(val > 0.f ? val : 0.f);
        }
    }
    __syncthreads();
    for (int idx = tid; idx < 64 * 64; idx += 256) {  // maxpool /2 over time
        const int tr = idx >> 6, c = idx & 63;
        const _Float16 a0 = Cs[(2 * tr) * 64 + c], a1 = Cs[(2 * tr + 1) * 64 + c];
        h1[(b * 128 + (t0 >> 1) + tr) * 64 + c] = (a0 > a1 ? a0 : a1);
    }
}

// ======================= Kernel 2: conv2 + relu + pool ======================
// h1:[512][128][64] f16, w2:[128][64][3], b2:[128] -> feats:[512][64][128] f32
// GEMM view: M=B*128 (tile 128 rows), K=192 (6 ksteps of 32), N=128.
__global__ __launch_bounds__(256) void k_conv2(const _Float16* __restrict__ h1,
                                               const float* __restrict__ w2,
                                               const float* __restrict__ b2,
                                               float* __restrict__ feats) {
    __shared__ _Float16 smem[128 * 32 + 32 * 128 + 128 * 128];  // 48 KB
    _Float16* As = smem;              // [128][32]
    _Float16* Bs = As + 128 * 32;     // [32][128]
    _Float16* Cs = Bs + 32 * 128;     // [128][128]
    const int tid = threadIdx.x;
    const int b = blockIdx.x;
    const int lane = tid & 31, wv = tid >> 5;
    v8f acc[8] = {};

    for (int s = 0; s < 6; ++s) {
        const int kb0 = s * 32;
        for (int idx = tid; idx < 128 * 32; idx += 256) {      // A stage (im2col)
            const int row = idx >> 5, kk = idx & 31;
            const int k = kb0 + kk, tap = k >> 6, c = k & 63;
            const int tt = row + tap - 1;
            As[row * 32 + kk] = (tt >= 0 && tt < 128) ? h1[(b * 128 + tt) * 64 + c] : (_Float16)0.f;
        }
        for (int idx = tid; idx < 32 * 128; idx += 256) {      // B stage (weights)
            const int kk = idx >> 7, n = idx & 127;
            const int k = kb0 + kk, tap = k >> 6, c = k & 63;
            Bs[kk * 128 + n] = (_Float16)w2[n * 192 + c * 3 + tap];
        }
        __syncthreads();
        const int mrow = wv * 16 + (lane & 15);
        v16h a = load_a_frag(As + mrow * 32, lane);
#pragma unroll
        for (int nt = 0; nt < 8; ++nt) {
            v16h bf = load_b_frag(Bs + nt * 16 + (lane & 15), 128, lane);
            acc[nt] = __builtin_amdgcn_wmma_f32_16x16x32_f16(false, a, false, bf,
                                                             (short)0, acc[nt], false, false);
        }
        __syncthreads();
    }
    const int colh = lane & 15, rh = lane >> 4;
#pragma unroll
    for (int nt = 0; nt < 8; ++nt) {
        const int c = nt * 16 + colh;
        const float bias = b2[c];
#pragma unroll
        for (int r = 0; r < 8; ++r) {
            const int m = wv * 16 + r + 8 * rh;
            const float val = acc[nt][r] + bias;
            Cs[m * 128 + c] = (_Float16)(val > 0.f ? val : 0.f);
        }
    }
    __syncthreads();
    for (int idx = tid; idx < 64 * 128; idx += 256) {  // maxpool /2
        const int tr = idx >> 7, j = idx & 127;
        const _Float16 a0 = Cs[(2 * tr) * 128 + j], a1 = Cs[(2 * tr + 1) * 128 + j];
        feats[(b * 64 + tr) * 128 + j] = (float)(a0 > a1 ? a0 : a1);
    }
}

// ======================= Kernel 3: LTC recurrence ==========================
// Recurrent matrices pinned in 256 KB of (dynamic) LDS; 4 batches per block.
__device__ __forceinline__ float sigmf(float z) { return 1.f / (1.f + __expf(-z)); }

__global__ __launch_bounds__(256) void k_lnn(const float* __restrict__ feats,
    const float* __restrict__ input_w, const float* __restrict__ input_b,
    const float* __restrict__ smu, const float* __restrict__ ssig,
    const float* __restrict__ sW,  const float* __restrict__ serev,
    const float* __restrict__ mu,  const float* __restrict__ sig,
    const float* __restrict__ W,   const float* __restrict__ erev,
    const float* __restrict__ vleak, const float* __restrict__ gleak,
    const float* __restrict__ cm,  float* __restrict__ hT) {
    extern __shared__ float sm[];
    float* Wl  = sm;               // 16384
    float* El  = Wl + 16384;
    float* Ml  = El + 16384;
    float* Sl  = Ml + 16384;
    float* vl  = Sl + 16384;       // [4][128]
    float* xl  = vl + 4 * 128;     // [4][128]
    float* iwl = xl + 4 * 128;     // 128
    float* ibl = iwl + 128;        // 128

    const int tid = threadIdx.x;        // 256
    const int j   = tid & 127;          // unit (column) owned by this thread
    const int sub = tid >> 7;           // 0/1 -> two batches each
    const int b0  = blockIdx.x * 4;

    for (int idx = tid; idx < 16384; idx += 256) {
        Wl[idx] = W[idx]; El[idx] = erev[idx]; Ml[idx] = mu[idx]; Sl[idx] = sig[idx];
    }
    if (tid < 128) { iwl[tid] = input_w[tid]; ibl[tid] = input_b[tid]; }
    for (int idx = tid; idx < 4 * 128; idx += 256) vl[idx] = 0.f;

    const float cmj = cm[j], glj = gleak[j];
    const float glvl = glj * vleak[j];
    float vreg[2] = {0.f, 0.f};
    __syncthreads();

    for (int t = 0; t < 64; ++t) {
        for (int idx = tid; idx < 4 * 128; idx += 256) {       // xt = x*iw + ib
            const int bb = idx >> 7, i = idx & 127;
            xl[idx] = feats[((b0 + bb) * 64 + t) * 128 + i] * iwl[i] + ibl[i];
        }
        __syncthreads();
        // ---- sensory pass (streams from L2; coalesced in j) ----
        float ns[2] = {0.f, 0.f}, ds[2] = {0.f, 0.f};
        const float* xb = xl + sub * 2 * 128;
        for (int i = 0; i < 128; ++i) {
            const int o = i * 128 + j;
            const float m_ = smu[o], s_ = ssig[o], w_ = sW[o], e_ = serev[o];
#pragma unroll
            for (int q = 0; q < 2; ++q) {
                const float a = w_ * sigmf((xb[q * 128 + i] - m_) * s_);
                ns[q] += a * e_; ds[q] += a;
            }
        }
        // ---- 6 semi-implicit unfolds (weights from LDS) ----
        const float* vb = vl + sub * 2 * 128;
        for (int u = 0; u < 6; ++u) {
            float wn[2] = {ns[0], ns[1]}, wd[2] = {ds[0], ds[1]};
            for (int i = 0; i < 128; ++i) {
                const int o = i * 128 + j;
                const float m_ = Ml[o], s_ = Sl[o], w_ = Wl[o], e_ = El[o];
#pragma unroll
                for (int q = 0; q < 2; ++q) {
                    const float a = w_ * sigmf((vb[q * 128 + i] - m_) * s_);
                    wn[q] += a * e_; wd[q] += a;
                }
            }
#pragma unroll
            for (int q = 0; q < 2; ++q)
                vreg[q] = (cmj * vreg[q] + glvl + wn[q]) / (cmj + glj + wd[q]);
            __syncthreads();                                   // all reads of old v done
#pragma unroll
            for (int q = 0; q < 2; ++q) vl[(sub * 2 + q) * 128 + j] = vreg[q];
            __syncthreads();
        }
    }
#pragma unroll
    for (int q = 0; q < 2; ++q) hT[(b0 + sub * 2 + q) * 128 + j] = vreg[q];
}

// ======================= Kernel 4: MLP head ================================
__global__ __launch_bounds__(64) void k_head(const float* __restrict__ hT,
                                             const float* __restrict__ fc1w,
                                             const float* __restrict__ fc1b,
                                             const float* __restrict__ fc2w,
                                             const float* __restrict__ fc2b,
                                             float* __restrict__ out) {
    __shared__ float red[64];
    const int b = blockIdx.x, d = threadIdx.x;
    const float* h = hT + b * 128;
    float acc = fc1b[d];
    for (int i = 0; i < 128; ++i) acc += fc1w[d * 128 + i] * h[i];
    acc = acc > 0.f ? acc : 0.f;
    red[d] = acc * fc2w[d];
    __syncthreads();
    for (int s = 32; s > 0; s >>= 1) { if (d < s) red[d] += red[d + s]; __syncthreads(); }
    if (d == 0) out[b] = red[0] + fc2b[0];
}

// ======================= Launch ============================================
extern "C" void kernel_launch(void* const* d_in, const int* in_sizes, int n_in,
                              void* d_out, int out_size, void* d_ws, size_t ws_size,
                              hipStream_t stream) {
    (void)in_sizes; (void)n_in; (void)out_size; (void)ws_size;
    const float* x       = (const float*)d_in[0];
    const float* conv1_w = (const float*)d_in[1];
    const float* conv1_b = (const float*)d_in[2];
    const float* conv2_w = (const float*)d_in[3];
    const float* conv2_b = (const float*)d_in[4];
    const float* input_w = (const float*)d_in[5];
    const float* input_b = (const float*)d_in[6];
    const float* smu     = (const float*)d_in[7];
    const float* ssig    = (const float*)d_in[8];
    const float* sW      = (const float*)d_in[9];
    const float* serev   = (const float*)d_in[10];
    const float* mu      = (const float*)d_in[11];
    const float* sig     = (const float*)d_in[12];
    const float* W       = (const float*)d_in[13];
    const float* erev    = (const float*)d_in[14];
    const float* vleak   = (const float*)d_in[15];
    const float* gleak   = (const float*)d_in[16];
    const float* cm_t    = (const float*)d_in[17];
    const float* fc1_w   = (const float*)d_in[18];
    const float* fc1_b   = (const float*)d_in[19];
    const float* fc2_w   = (const float*)d_in[20];
    const float* fc2_b   = (const float*)d_in[21];
    float* out = (float*)d_out;

    char* ws = (char*)d_ws;
    _Float16* h1 = (_Float16*)ws;                                 // 512*128*64 f16 = 8 MB
    float* feats = (float*)(ws + 8388608);                        // 512*64*128 f32 = 16 MB
    float* hTbuf = (float*)(ws + 8388608 + 16777216);             // 512*128 f32

    constexpr size_t LNN_SMEM = (size_t)(4 * 16384 + 2 * 4 * 128 + 2 * 128) * sizeof(float); // ~261 KB
    (void)hipFuncSetAttribute(reinterpret_cast<const void*>(k_lnn),
                              hipFuncAttributeMaxDynamicSharedMemorySize, (int)LNN_SMEM);

    k_conv1<<<1024, 256, 0, stream>>>(x, conv1_w, conv1_b, h1);
    k_conv2<<<512, 256, 0, stream>>>(h1, conv2_w, conv2_b, feats);
    k_lnn<<<128, 256, LNN_SMEM, stream>>>(feats, input_w, input_b,
                                          smu, ssig, sW, serev,
                                          mu, sig, W, erev,
                                          vleak, gleak, cm_t, hTbuf);
    k_head<<<512, 64, 0, stream>>>(hTbuf, fc1_w, fc1_b, fc2_w, fc2_b, out);
}
